// FeatExtractor_91199335563482
// MI455X (gfx1250) — matrix operand order repaired and checked by reading, not compiled, not run on hardware
//
#include <hip/hip_runtime.h>
#include <hip/hip_bf16.h>

// ---------------------------------------------------------------------------
// CDNA5 (gfx1250) fused GNN feature extractor.
// All GEMMs run on v_wmma_f32_16x16x32_bf16 (wave32, fp32 accumulate).
// BatchNorm batch statistics are fused into the GEMM passes via LDS reduce +
// one global atomic per channel per block; BN is then applied as a folded
// per-channel affine (a*x+s) in the *next* pass's LDS staging.
// Edge-level kernels (E % 128 == 0) compile with FULL=true -> no row guards,
// no exec-mask branch chains. bf16 activation staging is vectorized to
// global_load_b128 / ds_store_b128.
// ---------------------------------------------------------------------------

typedef __attribute__((ext_vector_type(16))) __bf16 v16bf;
typedef __attribute__((ext_vector_type(8)))  __bf16 v8bf;
typedef __attribute__((ext_vector_type(8)))  float  v8f;

#define N_NODES 50000
#define N_EDGES 800000

__device__ __forceinline__ float sigmoidf_(float x) { return 1.f / (1.f + __expf(-x)); }
__device__ __forceinline__ float siluf_(float x)    { return x * sigmoidf_(x); }
__device__ __forceinline__ float softplusf_(float x){ return fmaxf(x, 0.f) + log1pf(__expf(-fabsf(x))); }

// --- WMMA fragment loads from LDS (CDNA5 16-bit layouts, wave32) -----------
// A 16x32 (MxK): lane half=l>>4 holds K {half*16 .. +16} (contiguous)
__device__ __forceinline__ v16bf load_frag_a(const __bf16* rowKc, int half) {
  const __bf16* p = rowKc + half * 16;
  v8bf lo = *(const v8bf*)(p);
  v8bf hi = *(const v8bf*)(p + 8);
  v16bf r;
#pragma unroll
  for (int e = 0; e < 8; ++e) { r[e] = lo[e]; r[e + 8] = hi[e]; }
  return r;
}
// B 32x16 (KxN) staged transposed in LDS (row = n, contiguous k):
// lane half holds k = 16*half + e for column n = lane&15
__device__ __forceinline__ v16bf load_frag_b(const __bf16* rowKc, int half) {
  const __bf16* p = rowKc + half * 16;
  v8bf lo = *(const v8bf*)(p);
  v8bf hi = *(const v8bf*)(p + 8);
  v16bf r;
#pragma unroll
  for (int e = 0; e < 8; ++e) { r[e] = lo[e]; r[e + 8] = hi[e]; }
  return r;
}

// --- Pre-op: 0 = none, 1 = BN-fold affine, 2 = affine + SiLU ---------------
template <int PRE>
__device__ __forceinline__ float preop_(float v, int k,
                                        const float* __restrict__ preA,
                                        const float* __restrict__ preS) {
  if constexpr (PRE >= 1) v = preA[k] * v + preS[k];
  if constexpr (PRE == 2) v = siluf_(v);
  return v;
}

// --- Activation staging into LDS (bf16, pre-op fused) ----------------------
// bf16 source + KSRC==KP: vectorized 8-wide (b128 global load, b128 LDS store)
template <typename TIN, int KSRC, int KP, int PRE, bool FULL>
__device__ __forceinline__ void stage_A(const TIN* __restrict__ X, int row0, int M,
                                        const float* __restrict__ preA,
                                        const float* __restrict__ preS,
                                        __bf16* __restrict__ ldsA, int tid) {
  constexpr int LDK = KP + 8;
  if constexpr (sizeof(TIN) == 2 && KSRC == KP) {
    constexpr int KV = KP / 8;  // 8-element vectors per row
    for (int idx = tid; idx < 128 * KV; idx += 256) {
      int r = idx / KV, kv = idx - r * KV;
      int gr = row0 + r;
      int k0 = kv * 8;
      v8bf in;
      if (FULL || gr < M) {
        in = *(const v8bf*)(X + (size_t)gr * KSRC + k0);
        if (FULL || gr + 128 < M)  // prefetch next block-tile's rows
          __builtin_prefetch(X + (size_t)(gr + 128) * KSRC + k0, 0, 1);
      } else {
#pragma unroll
        for (int e = 0; e < 8; ++e) in[e] = (__bf16)0.f;
      }
      v8bf out;
#pragma unroll
      for (int e = 0; e < 8; ++e)
        out[e] = (__bf16)preop_<PRE>((float)in[e], k0 + e, preA, preS);
      *(v8bf*)(&ldsA[r * LDK + k0]) = out;
    }
  } else {
    for (int idx = tid; idx < 128 * KP; idx += 256) {
      int r = idx / KP, k = idx - r * KP;
      int gr = row0 + r;
      float v = 0.f;
      if ((FULL || gr < M) && k < KSRC)
        v = preop_<PRE>((float)X[(size_t)gr * KSRC + k], k, preA, preS);
      ldsA[r * LDK + k] = (__bf16)v;
    }
  }
}

// --- Weight staging: fp32 -> bf16, transposed, K zero-padded ---------------
template <int KSRC, int KP, int NOUT>
__device__ __forceinline__ void stage_B(const float* __restrict__ W,
                                        __bf16* __restrict__ ldsB, int tid) {
  constexpr int LDK = KP + 8;
  for (int idx = tid; idx < NOUT * KP; idx += 256) {
    int n = idx / KP, k = idx - n * KP;
    float v = (k < KSRC) ? W[(size_t)k * NOUT + n] : 0.f;
    ldsB[n * LDK + k] = (__bf16)v;
  }
}

// ---------------------------------------------------------------------------
// Generic fused layer:  Y = preop(X) @ W + bias   (+ column sum / sumsq)
//   Block = 256 threads = 8 waves; block tile = 128 rows x NOUT cols.
//   A (128 x KP) and W^T (NOUT x KP) staged whole-K in LDS, bf16.
// ---------------------------------------------------------------------------
template <typename TIN, typename TOUT, int KSRC, int KP, int NOUT, int PRE,
          bool STORE, bool STATS, bool FULL>
__global__ __launch_bounds__(256) void layer_gemm(
    const TIN* __restrict__ X, int M,
    const float* __restrict__ W, const float* __restrict__ bias,
    const float* __restrict__ preA, const float* __restrict__ preS,
    TOUT* __restrict__ Y,
    float* __restrict__ gSum, float* __restrict__ gSq) {
  constexpr int LDK = KP + 8;  // pad: keeps 16B alignment, kills bank conflicts
  __shared__ __bf16 ldsA[128 * LDK];
  __shared__ __bf16 ldsB[NOUT * LDK];
  __shared__ float sSum[NOUT];
  __shared__ float sSq[NOUT];

  const int tid  = threadIdx.x;
  const int row0 = blockIdx.x * 128;

  if (STATS) {
    for (int i = tid; i < NOUT; i += 256) { sSum[i] = 0.f; sSq[i] = 0.f; }
  }
  stage_A<TIN, KSRC, KP, PRE, FULL>(X, row0, M, preA, preS, ldsA, tid);
  stage_B<KSRC, KP, NOUT>(W, ldsB, tid);
  __syncthreads();

  const int wave = tid >> 5, lane = tid & 31;
  const int half = lane >> 4, ln = lane & 15;
  const __bf16* aRow = &ldsA[(wave * 16 + ln) * LDK];

#pragma unroll 1
  for (int j = 0; j < NOUT / 16; ++j) {
    v8f acc;
#pragma unroll
    for (int r = 0; r < 8; ++r) acc[r] = 0.f;
    const __bf16* bRow = &ldsB[(j * 16 + ln) * LDK];
#pragma unroll
    for (int kc = 0; kc < KP; kc += 32) {
      v16bf af = load_frag_a(aRow + kc, half);
      v16bf bf = load_frag_b(bRow + kc, half);
      acc = __builtin_amdgcn_wmma_f32_16x16x32_bf16(false, af, false, bf,
                                                    (short)0, acc, false, false);
    }
    const int n = j * 16 + ln;
    const float bn = bias[n];
    float ps = 0.f, pq = 0.f;
#pragma unroll
    for (int r = 0; r < 8; ++r) {
      int gr = row0 + wave * 16 + r + 8 * half;
      if (FULL || gr < M) {
        float y = acc[r] + bn;
        if constexpr (STORE) Y[(size_t)gr * NOUT + n] = (TOUT)y;
        ps += y; pq += y * y;
      }
    }
    if (STATS) { atomicAdd(&sSum[n], ps); atomicAdd(&sSq[n], pq); }
  }
  if (STATS) {
    __syncthreads();
    for (int i = tid; i < NOUT; i += 256) {
      atomicAdd(&gSum[i], sSum[i]);
      atomicAdd(&gSq[i], sSq[i]);
    }
  }
}

// ---------------------------------------------------------------------------
// chem layer-2 recompute + gate + segment_sum scatter (edge rows, always full
// tiles: E % 128 == 0 -> unconditional, branch-free atomics).
// Computes filter tile (cols 16j..) and core tile (cols 128+16j..) together;
// g = sigmoid(BN(filter)) * softplus(BN(core)), atomic-add into h_chem[vid]
// (25.6 MB table, L2-resident on the 192 MB L2).
// ---------------------------------------------------------------------------
__global__ __launch_bounds__(256) void chem2_scatter(
    const __bf16* __restrict__ X1,
    const float* __restrict__ W, const float* __restrict__ bias,
    const float* __restrict__ p1A, const float* __restrict__ p1S,
    const float* __restrict__ f2A, const float* __restrict__ f2S,
    const int* __restrict__ nbr, float* __restrict__ hchem) {
  constexpr int KP = 128, NOUT = 256, LDK = KP + 8;
  __shared__ __bf16 ldsA[128 * LDK];
  __shared__ __bf16 ldsB[NOUT * LDK];
  const int tid = threadIdx.x;
  const int row0 = blockIdx.x * 128;

  stage_A<__bf16, KP, KP, 2, true>(X1, row0, N_EDGES, p1A, p1S, ldsA, tid);
  stage_B<KP, KP, NOUT>(W, ldsB, tid);
  __syncthreads();

  const int wave = tid >> 5, lane = tid & 31;
  const int half = lane >> 4, ln = lane & 15;
  const __bf16* aRow = &ldsA[(wave * 16 + ln) * LDK];

  int vids[8];
#pragma unroll
  for (int r = 0; r < 8; ++r)
    vids[r] = nbr[row0 + wave * 16 + r + 8 * half];

#pragma unroll 1
  for (int j = 0; j < 8; ++j) {
    v8f aF, aC;
#pragma unroll
    for (int r = 0; r < 8; ++r) { aF[r] = 0.f; aC[r] = 0.f; }
    const __bf16* bRowF = &ldsB[(j * 16 + ln) * LDK];
    const __bf16* bRowC = &ldsB[((j + 8) * 16 + ln) * LDK];
#pragma unroll
    for (int kc = 0; kc < KP; kc += 32) {
      v16bf af  = load_frag_a(aRow + kc, half);
      v16bf bfF = load_frag_b(bRowF + kc, half);
      aF = __builtin_amdgcn_wmma_f32_16x16x32_bf16(false, af, false, bfF,
                                                   (short)0, aF, false, false);
      v16bf bfC = load_frag_b(bRowC + kc, half);
      aC = __builtin_amdgcn_wmma_f32_16x16x32_bf16(false, af, false, bfC,
                                                   (short)0, aC, false, false);
    }
    const int nF = j * 16 + ln, nC = 128 + j * 16 + ln;
    const float bF = bias[nF], bC = bias[nC];
    const float a2F = f2A[nF], s2F = f2S[nF];
    const float a2C = f2A[nC], s2C = f2S[nC];
#pragma unroll
    for (int r = 0; r < 8; ++r) {
      float yF = a2F * (aF[r] + bF) + s2F;
      float yC = a2C * (aC[r] + bC) + s2C;
      float g  = sigmoidf_(yF) * softplusf_(yC);
      atomicAdd(&hchem[(size_t)vids[r] * 128 + nF], g);
    }
  }
}

// --- BN stats -> folded affine (a, s):  y = a*x + s ------------------------
__global__ void bn_finalize(const float* __restrict__ s, const float* __restrict__ q,
                            const float* __restrict__ gamma, const float* __restrict__ beta,
                            float invM, int C,
                            float* __restrict__ fA, float* __restrict__ fS) {
  int c = blockIdx.x * blockDim.x + threadIdx.x;
  if (c < C) {
    float m = s[c] * invM;
    float v = q[c] * invM - m * m;
    float inv = rsqrtf(v + 1e-5f);
    float a = gamma[c] * inv;
    fA[c] = a;
    fS[c] = beta[c] - m * a;
  }
}

// --- concat(h_chem, BN(geom2)) -> bf16 [N,192] -----------------------------
__global__ void build_fuse(const float* __restrict__ hchem, const float* __restrict__ yg2,
                           const float* __restrict__ gA, const float* __restrict__ gS,
                           __bf16* __restrict__ xf) {
  int idx = blockIdx.x * 256 + threadIdx.x;
  if (idx < N_NODES * 192) {
    int i = idx / 192, c = idx - i * 192;
    float v;
    if (c < 128) v = hchem[(size_t)i * 128 + c];
    else { int cc = c - 128; v = gA[cc] * yg2[(size_t)i * 64 + cc] + gS[cc]; }
    xf[idx] = (__bf16)v;
  }
}

// --- final BN apply -> d_out ----------------------------------------------
__global__ void final_apply(const float* __restrict__ y,
                            const float* __restrict__ fA, const float* __restrict__ fS,
                            float* __restrict__ out) {
  int idx = blockIdx.x * 256 + threadIdx.x;
  if (idx < N_NODES * 128) { int c = idx & 127; out[idx] = fA[c] * y[idx] + fS[c]; }
}

// ---------------------------------------------------------------------------
extern "C" void kernel_launch(void* const* d_in, const int* in_sizes, int n_in,
                              void* d_out, int out_size, void* d_ws, size_t ws_size,
                              hipStream_t stream) {
  const float* chem = (const float*)d_in[0];
  const float* geom = (const float*)d_in[1];
  const int*   nbr  = (const int*)d_in[2];
  const float *w_c1 = (const float*)d_in[3],  *b_c1 = (const float*)d_in[4];
  const float *g_c1 = (const float*)d_in[5],  *be_c1 = (const float*)d_in[6];
  const float *w_c2 = (const float*)d_in[7],  *b_c2 = (const float*)d_in[8];
  const float *g_c2 = (const float*)d_in[9],  *be_c2 = (const float*)d_in[10];
  const float *w_g1 = (const float*)d_in[11], *b_g1 = (const float*)d_in[12];
  const float *g_g1 = (const float*)d_in[13], *be_g1 = (const float*)d_in[14];
  const float *w_g2 = (const float*)d_in[15], *b_g2 = (const float*)d_in[16];
  const float *g_g2 = (const float*)d_in[17], *be_g2 = (const float*)d_in[18];
  const float *w_f1 = (const float*)d_in[19], *b_f1 = (const float*)d_in[20];
  const float *g_f1 = (const float*)d_in[21], *be_f1 = (const float*)d_in[22];
  const float *w_f2 = (const float*)d_in[23], *b_f2 = (const float*)d_in[24];
  const float *g_f2 = (const float*)d_in[25], *be_f2 = (const float*)d_in[26];

  char* p = (char*)d_ws;
  auto alloc = [&](size_t n) { void* r = (void*)p; p += (n + 255) & ~(size_t)255; return r; };

  __bf16* X1    = (__bf16*)alloc((size_t)N_EDGES * 128 * sizeof(__bf16)); // 205 MB
  float*  hchem = (float*) alloc((size_t)N_NODES * 128 * sizeof(float));  // 25.6 MB
  __bf16* Xg1   = (__bf16*)alloc((size_t)N_NODES * 64 * sizeof(__bf16));
  float*  Yg2   = (float*) alloc((size_t)N_NODES * 64 * sizeof(float));
  __bf16* Xf    = (__bf16*)alloc((size_t)N_NODES * 192 * sizeof(__bf16));
  __bf16* Yf1   = (__bf16*)alloc((size_t)N_NODES * 128 * sizeof(__bf16));
  float*  Yf2   = (float*) alloc((size_t)N_NODES * 128 * sizeof(float));
  float*  st    = (float*) alloc(1536 * sizeof(float));  // sums+sumsqs all layers
  float*  fo    = (float*) alloc(1536 * sizeof(float));  // folded a,s all layers

  // stats/fold layout (floats): c1[128x2] c2[256x2] g1[64x2] g2[64x2] f1[128x2] f2[128x2]
  float *sum_c1 = st,        *sq_c1 = st + 128;
  float *sum_c2 = st + 256,  *sq_c2 = st + 512;
  float *sum_g1 = st + 768,  *sq_g1 = st + 832;
  float *sum_g2 = st + 896,  *sq_g2 = st + 960;
  float *sum_f1 = st + 1024, *sq_f1 = st + 1152;
  float *sum_f2 = st + 1280, *sq_f2 = st + 1408;
  float *fA_c1 = fo,        *fS_c1 = fo + 128;
  float *fA_c2 = fo + 256,  *fS_c2 = fo + 512;
  float *fA_g1 = fo + 768,  *fS_g1 = fo + 832;
  float *fA_g2 = fo + 896,  *fS_g2 = fo + 960;
  float *fA_f1 = fo + 1024, *fS_f1 = fo + 1152;
  float *fA_f2 = fo + 1280, *fS_f2 = fo + 1408;

  hipMemsetAsync(st, 0, 1536 * sizeof(float), stream);
  hipMemsetAsync(hchem, 0, (size_t)N_NODES * 128 * sizeof(float), stream);

  dim3 B(256);
  const int gbE = N_EDGES / 128;            // 6250 (exact -> FULL tiles)
  const int gbN = (N_NODES + 127) / 128;    // 391  (tail -> guarded)
  const float invE = 1.f / (float)N_EDGES;
  const float invN = 1.f / (float)N_NODES;

  // ---- chem branch (edge-level, WMMA-heavy, branch-free) ----
  layer_gemm<float, __bf16, 34, 64, 128, 0, true, true, true>
      <<<gbE, B, 0, stream>>>(chem, N_EDGES, w_c1, b_c1, nullptr, nullptr,
                              X1, sum_c1, sq_c1);
  bn_finalize<<<1, 128, 0, stream>>>(sum_c1, sq_c1, g_c1, be_c1, invE, 128, fA_c1, fS_c1);

  layer_gemm<__bf16, float, 128, 128, 256, 2, false, true, true>
      <<<gbE, B, 0, stream>>>(X1, N_EDGES, w_c2, b_c2, fA_c1, fS_c1,
                              (float*)nullptr, sum_c2, sq_c2);
  bn_finalize<<<1, 256, 0, stream>>>(sum_c2, sq_c2, g_c2, be_c2, invE, 256, fA_c2, fS_c2);

  chem2_scatter<<<gbE, B, 0, stream>>>(X1, w_c2, b_c2, fA_c1, fS_c1,
                                       fA_c2, fS_c2, nbr, hchem);

  // ---- geom branch (node-level) ----
  layer_gemm<float, __bf16, 48, 64, 64, 0, true, true, false>
      <<<gbN, B, 0, stream>>>(geom, N_NODES, w_g1, b_g1, nullptr, nullptr,
                              Xg1, sum_g1, sq_g1);
  bn_finalize<<<1, 64, 0, stream>>>(sum_g1, sq_g1, g_g1, be_g1, invN, 64, fA_g1, fS_g1);

  layer_gemm<__bf16, float, 64, 64, 64, 2, true, true, false>
      <<<gbN, B, 0, stream>>>(Xg1, N_NODES, w_g2, b_g2, fA_g1, fS_g1,
                              Yg2, sum_g2, sq_g2);
  bn_finalize<<<1, 64, 0, stream>>>(sum_g2, sq_g2, g_g2, be_g2, invN, 64, fA_g2, fS_g2);

  // ---- fuse ----
  build_fuse<<<(N_NODES * 192 + 255) / 256, B, 0, stream>>>(hchem, Yg2, fA_g2, fS_g2, Xf);

  layer_gemm<__bf16, __bf16, 192, 192, 128, 0, true, true, false>
      <<<gbN, B, 0, stream>>>(Xf, N_NODES, w_f1, b_f1, nullptr, nullptr,
                              Yf1, sum_f1, sq_f1);
  bn_finalize<<<1, 128, 0, stream>>>(sum_f1, sq_f1, g_f1, be_f1, invN, 128, fA_f1, fS_f1);

  layer_gemm<__bf16, float, 128, 128, 128, 2, true, true, false>
      <<<gbN, B, 0, stream>>>(Yf1, N_NODES, w_f2, b_f2, fA_f1, fS_f1,
                              Yf2, sum_f2, sq_f2);
  bn_finalize<<<1, 128, 0, stream>>>(sum_f2, sq_f2, g_f2, be_f2, invN, 128, fA_f2, fS_f2);

  final_apply<<<(N_NODES * 128 + 255) / 256, B, 0, stream>>>(Yf2, fA_f2, fS_f2, (float*)d_out);
}
